// DynamicsEnsemble_13365938225568
// MI455X (gfx1250) — compile-verified, matrix operand order checked by reading, not run on hardware
//
#include <hip/hip_runtime.h>
#include <hip/hip_bf16.h>

// Problem constants (from reference)
#define Bsz   8192
#define Sdim  256
#define Adim  64
#define Hdim  1024
#define Edim  8
#define INdim 320      // S + A
#define OUTdim 257     // S + 1
#define W3P   384      // W3 rows (N) padded to 3 * BN in transposed layout

// Tile config
#define BM 128
#define BN 128
#define BK 32

typedef __bf16 bf16;
typedef bf16  v16bf __attribute__((ext_vector_type(16)));
typedef bf16  v8bf  __attribute__((ext_vector_type(8)));
typedef float v8f   __attribute__((ext_vector_type(8)));
typedef unsigned int u32x4 __attribute__((ext_vector_type(4)));
typedef int          i32x8 __attribute__((ext_vector_type(8)));
typedef int          i32x4 __attribute__((ext_vector_type(4)));

__device__ __forceinline__ void split32(float v, bf16* hi, bf16* lo) {
    bf16 h = (bf16)v;
    *hi = h;
    *lo = (bf16)(v - (float)h);
}

// ---------------- workspace init: zero counts, fill rowlist with -1 ------------
__global__ void dyn_init_kernel(int* counts, int* rowlist, int n) {
    int i = blockIdx.x * blockDim.x + threadIdx.x;
    if (i < n) rowlist[i] = -1;
    if (i < Edim) counts[i] = 0;
}

// ---------------- bucket rows by expert ---------------------------------------
__global__ void dyn_bucket_kernel(const int* __restrict__ idx,
                                  int* counts, int* rowlist) {
    int i = blockIdx.x * blockDim.x + threadIdx.x;
    if (i < Bsz) {
        int e = idx[i];
        int p = atomicAdd(&counts[e], 1);
        rowlist[e * Bsz + p] = i;
    }
}

// ---- transpose + split: W[e][k][n] (fp32) -> WT[e][n][k] (bf16 hi/lo) --------
// K-contiguous rows so GEMM B-tiles are dense 2D tiles (TDM-friendly).
// block (32,8); grid (K/32, Np/32, E). Pads n in [Nsrc, Np) with zeros.
__global__ void dyn_tsplit_kernel(const float* __restrict__ src,
                                  bf16* __restrict__ hi, bf16* __restrict__ lo,
                                  int K, int Nsrc, int Np) {
    __shared__ float t[32][33];
    int e  = blockIdx.z;
    int k0 = blockIdx.x * 32;
    int n0 = blockIdx.y * 32;
    int tx = threadIdx.x, ty = threadIdx.y;
#pragma unroll
    for (int j = 0; j < 4; ++j) {
        int row = k0 + ty + j * 8;     // k (always < K: K multiple of 32)
        int col = n0 + tx;             // n
        float v = 0.f;
        if (col < Nsrc) v = src[((long)e * K + row) * Nsrc + col];
        t[ty + j * 8][tx] = v;
    }
    __syncthreads();
#pragma unroll
    for (int j = 0; j < 4; ++j) {
        int n = n0 + ty + j * 8;
        int k = k0 + tx;
        long o = ((long)e * Np + n) * K + k;
        split32(t[tx][ty + j * 8], &hi[o], &lo[o]);
    }
}

// Build a 16-element WMMA fragment from two contiguous 8-element LDS chunks.
// ISA 16-bit A/B layout: lane half selects kbase in {0,8}; element i maps to
// k = kbase + (i&7) + (i>>3)*16, i.e. chunks [kbase, kbase+8) and [kbase+16, kbase+24).
__device__ __forceinline__ v16bf frag16(const bf16* rowbase, int kbase) {
    v8bf c0 = *(const v8bf*)(rowbase + kbase);
    v8bf c1 = *(const v8bf*)(rowbase + kbase + 16);
    return __builtin_shufflevector(c0, c1, 0, 1, 2, 3, 4, 5, 6, 7,
                                           8, 9, 10, 11, 12, 13, 14, 15);
}

// ---------------- fused expert-bucketed GEMM (3-term bf16 split) --------------
// B tiles: Tensor Data Mover DMA (double-buffered LDS, pad_enable gives the
//          80-byte row pitch), issued by wave 0, ordered by s_wait_tensorcnt.
// A tiles: register-staged software pipeline (gathered rows; TDM can't gather).
template <int L>
__global__ __launch_bounds__(256)
void dyn_gemm_kernel(const int* __restrict__ counts, const int* __restrict__ rowlist,
                     const float* __restrict__ state, const float* __restrict__ action,
                     const bf16* __restrict__ Ahi, const bf16* __restrict__ Alo,
                     const bf16* __restrict__ Whi, const bf16* __restrict__ Wlo,
                     int Nn /*WT rows per expert*/, int Kdim,
                     const float* __restrict__ bias,
                     bf16* __restrict__ Ohi, bf16* __restrict__ Olo,
                     float* __restrict__ dnext, float* __restrict__ drew) {
    const int e   = blockIdx.z;
    const int cnt = counts[e];
    const int m0  = blockIdx.x * BM;
    if (m0 >= cnt) return;                 // block-uniform: EXEC stays all-1s
    const int n0  = blockIdx.y * BN;

    __shared__ int rowIdx[BM];
    __shared__ __align__(16) bf16 sAhi[BM][BK + 8], sAlo[BM][BK + 8];
    __shared__ __align__(16) bf16 sBThi[2][BN][BK + 8];   // double-buffered, TDM dest
    __shared__ __align__(16) bf16 sBTlo[2][BN][BK + 8];

    const int tid = threadIdx.x;
    for (int m = tid; m < BM; m += 256) {
        int g = m0 + m;
        rowIdx[m] = (g < cnt) ? rowlist[e * Bsz + g] : -1;
    }
    __syncthreads();

    const int lane  = tid & 31;
    const int wid   = tid >> 5;
    const int waveM = wid & 3;             // 4 waves over M (32 rows each)
    const int waveN = wid >> 2;            // 2 waves over N (64 cols each)
    const int laneR = lane & 15;
    const int kbase = (lane >> 4) * 8;     // ISA 16-bit A/B lane-half K offset

    v8f acc[2][4] = {};

    const int kSteps = Kdim / BK;

    // ---- TDM: DMA one (BN x BK) weight tile into an LDS buffer --------------
    // D# group0: count=1, lds_addr, 57-bit global_addr, type=2.
    // D# group1: data_size=2B, pad_enable, pad_interval=16 DWORDs (one 64B row),
    //            pad_amount=4 DWORDs (16B) -> 80B LDS pitch == BK+8 elements.
    //            tensor_dim0=Kdim, tensor_dim1=Nn, tile=32x128, stride0=Kdim.
    auto issueTdmB = [&](int kb, int buf) {
        if (wid != 0) return;              // one DMA pair per workgroup
        long elemOff = ((long)e * Nn + n0) * (long)Kdim + (long)kb * BK;
#pragma unroll
        for (int hl = 0; hl < 2; ++hl) {
            const bf16* srcp = hl ? (Wlo + elemOff) : (Whi + elemOff);
            unsigned long long ga = (unsigned long long)(const void*)srcp;
            bf16* ldsp = hl ? &sBTlo[buf][0][0] : &sBThi[buf][0][0];
            unsigned ldsa = (unsigned)(unsigned long long)(void*)ldsp;
            u32x4 g0 = { 1u,                                  // count=1 (valid)
                         ldsa,                                // lds_addr
                         (unsigned)ga,                        // global_addr[31:0]
                         (unsigned)((ga >> 32) & 0x01FFFFFFull) | (2u << 30) };
            i32x8 g1 = { (int)((1u << 16) | (1u << 20) | (3u << 22) | (3u << 25)),
                         (int)(((unsigned)Kdim & 0xFFFFu) << 16),   // tdim0 lo
                         (int)(((unsigned)Nn & 0xFFFFu) << 16),     // tdim1 lo
                         (int)(32u << 16),                          // tile_dim0=32
                         128,                                       // tile_dim1=128
                         Kdim,                                      // stride0
                         0, 0 };
            i32x4 gz4 = { 0, 0, 0, 0 };
            i32x8 gz8 = { 0, 0, 0, 0, 0, 0, 0, 0 };
            __builtin_amdgcn_tensor_load_to_lds(g0, g1, gz4, gz4, gz8, 0);
        }
    };

    // ---- A side: register staging (2 fixed chunks per thread) ---------------
    v8bf  rAh[2], rAl[2];
    float rAf[2][8];                       // L1 only (dead-code eliminated otherwise)

    auto globalLoadA = [&](int kb) {
#pragma unroll
        for (int c = 0; c < 2; ++c) {
            int i = tid + c * 256;
            int m = i >> 2, kc = (i & 3) * 8;
            int r = rowIdx[m];
            int k = kb * BK + kc;
            if constexpr (L == 1) {
#pragma unroll
                for (int j = 0; j < 8; ++j) rAf[c][j] = 0.f;
                if (r >= 0) {
                    // 8-aligned chunk; Sdim=256 is a multiple of 8 so a chunk
                    // never straddles the state/action boundary
                    const float* src = (k < Sdim)
                        ? (state + (long)r * Sdim + k)
                        : (action + (long)r * Adim + (k - Sdim));
#pragma unroll
                    for (int j = 0; j < 8; ++j) rAf[c][j] = src[j];
                }
            } else {
                rAh[c] = {}; rAl[c] = {};
                if (r >= 0) {
                    long o = (long)r * Hdim + k;
                    rAh[c] = *(const v8bf*)(Ahi + o);
                    rAl[c] = *(const v8bf*)(Alo + o);
                }
            }
        }
    };
    auto ldsCommitA = [&]() {
#pragma unroll
        for (int c = 0; c < 2; ++c) {
            int i = tid + c * 256;
            int m = i >> 2, kc = (i & 3) * 8;
            if constexpr (L == 1) {
                v8bf hv = {}, lv = {};
#pragma unroll
                for (int j = 0; j < 8; ++j) {
                    bf16 h = (bf16)rAf[c][j];
                    hv[j] = h;
                    lv[j] = (bf16)(rAf[c][j] - (float)h);
                }
                *(v8bf*)&sAhi[m][kc] = hv;
                *(v8bf*)&sAlo[m][kc] = lv;
            } else {
                *(v8bf*)&sAhi[m][kc] = rAh[c];
                *(v8bf*)&sAlo[m][kc] = rAl[c];
            }
        }
    };

    // pipeline prologue: stage tile 0
    issueTdmB(0, 0);
    globalLoadA(0);
    ldsCommitA();
    if (wid == 0) __builtin_amdgcn_s_wait_tensorcnt((short)0);
    __syncthreads();

    for (int kb = 0; kb < kSteps; ++kb) {
        const bool more = (kb + 1 < kSteps);   // block-uniform
        if (more) {
            issueTdmB(kb + 1, (kb + 1) & 1);   // DMA next B tile into other buffer
            globalLoadA(kb + 1);               // overlaps with WMMAs below
        }
        const int buf = kb & 1;

        // ---- fragments: contiguous ds_load_b128 pairs ----
        v16bf aH[2], aL[2], bH[4], bL[4];
#pragma unroll
        for (int ni = 0; ni < 4; ++ni) {
            const bf16* cb = &sBThi[buf][waveN * 64 + ni * 16 + laneR][0];
            const bf16* cl = &sBTlo[buf][waveN * 64 + ni * 16 + laneR][0];
            bH[ni] = frag16(cb, kbase);
            bL[ni] = frag16(cl, kbase);
        }
#pragma unroll
        for (int mi = 0; mi < 2; ++mi) {
            const bf16* rb = &sAhi[waveM * 32 + mi * 16 + laneR][0];
            const bf16* rl = &sAlo[waveM * 32 + mi * 16 + laneR][0];
            aH[mi] = frag16(rb, kbase);
            aL[mi] = frag16(rl, kbase);
        }

        // ---- 3-term compensated bf16 WMMA: hi*hi + hi*lo + lo*hi ----
#pragma unroll
        for (int mi = 0; mi < 2; ++mi)
#pragma unroll
            for (int ni = 0; ni < 4; ++ni) {
                acc[mi][ni] = __builtin_amdgcn_wmma_f32_16x16x32_bf16(
                    false, aH[mi], false, bH[ni], (short)0, acc[mi][ni], false, false);
                acc[mi][ni] = __builtin_amdgcn_wmma_f32_16x16x32_bf16(
                    false, aH[mi], false, bL[ni], (short)0, acc[mi][ni], false, false);
                acc[mi][ni] = __builtin_amdgcn_wmma_f32_16x16x32_bf16(
                    false, aL[mi], false, bH[ni], (short)0, acc[mi][ni], false, false);
            }

        __syncthreads();                       // everyone done reading this tile
        if (more) {
            ldsCommitA();                      // commit A tile kb+1 (waits loads)
            if (wid == 0) __builtin_amdgcn_s_wait_tensorcnt((short)0);
            __syncthreads();
        }
    }

    // ---- epilogue (C layout: row = 8*(lane>>4)+j, col = lane&15) ----
#pragma unroll
    for (int mi = 0; mi < 2; ++mi)
#pragma unroll
        for (int ni = 0; ni < 4; ++ni)
#pragma unroll
            for (int j = 0; j < 8; ++j) {
                int row = waveM * 32 + mi * 16 + ((lane >> 4) << 3) + j;
                int col = n0 + waveN * 64 + ni * 16 + laneR;
                int r   = rowIdx[row];
                if (r < 0) continue;
                float v = acc[mi][ni][j];
                if constexpr (L == 3) {
                    if (col < OUTdim) {
                        v += bias[e * OUTdim + col];
                        if (col < Sdim)
                            dnext[(long)r * Sdim + col] = state[(long)r * Sdim + col] + v;
                        else
                            drew[r] = v;
                    }
                } else {
                    v += bias[e * Hdim + col];
                    v = v > 0.f ? v : 0.f;
                    long o = (long)r * Hdim + col;
                    split32(v, &Ohi[o], &Olo[o]);
                }
            }
}

extern "C" void kernel_launch(void* const* d_in, const int* in_sizes, int n_in,
                              void* d_out, int out_size, void* d_ws, size_t ws_size,
                              hipStream_t stream) {
    const float* state  = (const float*)d_in[0];
    const float* action = (const float*)d_in[1];
    const int*   idx    = (const int*)d_in[2];
    const float* W1     = (const float*)d_in[3];
    const float* b1     = (const float*)d_in[4];
    const float* W2     = (const float*)d_in[5];
    const float* b2     = (const float*)d_in[6];
    const float* W3     = (const float*)d_in[7];
    const float* b3     = (const float*)d_in[8];

    float* dnext = (float*)d_out;                       // (B, 256)
    float* drew  = (float*)d_out + (long)Bsz * Sdim;    // (B, 1)

    // workspace carve-out
    char*  ws  = (char*)d_ws;
    size_t off = 0;
    auto carve = [&](size_t bytes) {
        void* p = ws + off;
        off = (off + bytes + 255) & ~(size_t)255;
        return p;
    };
    int*  counts  = (int*)carve(Edim * sizeof(int));
    int*  rowlist = (int*)carve((size_t)Edim * Bsz * sizeof(int));
    // transposed weights WT[e][n][k], bf16 hi/lo
    bf16* W1hi = (bf16*)carve((size_t)Edim * Hdim * INdim * 2);
    bf16* W1lo = (bf16*)carve((size_t)Edim * Hdim * INdim * 2);
    bf16* W2hi = (bf16*)carve((size_t)Edim * Hdim * Hdim * 2);
    bf16* W2lo = (bf16*)carve((size_t)Edim * Hdim * Hdim * 2);
    bf16* W3hi = (bf16*)carve((size_t)Edim * W3P * Hdim * 2);
    bf16* W3lo = (bf16*)carve((size_t)Edim * W3P * Hdim * 2);
    bf16* h1hi = (bf16*)carve((size_t)Bsz * Hdim * 2);
    bf16* h1lo = (bf16*)carve((size_t)Bsz * Hdim * 2);
    bf16* h2hi = (bf16*)carve((size_t)Bsz * Hdim * 2);
    bf16* h2lo = (bf16*)carve((size_t)Bsz * Hdim * 2);
    (void)ws_size; (void)in_sizes; (void)n_in; (void)out_size;

    // 1) init + bucket rows by expert
    {
        int n = Edim * Bsz;
        dyn_init_kernel<<<(n + 255) / 256, 256, 0, stream>>>(counts, rowlist, n);
        dyn_bucket_kernel<<<(Bsz + 255) / 256, 256, 0, stream>>>(idx, counts, rowlist);
    }
    // 2) transpose + split weights -> WT[e][n][k] bf16 hi/lo (K-contiguous rows)
    {
        dim3 tb(32, 8, 1);
        dim3 gw1(INdim / 32, Hdim / 32, Edim);   // (10, 32, 8)
        dim3 gw2(Hdim / 32,  Hdim / 32, Edim);   // (32, 32, 8)
        dim3 gw3(Hdim / 32,  W3P  / 32, Edim);   // (32, 12, 8)
        dyn_tsplit_kernel<<<gw1, tb, 0, stream>>>(W1, W1hi, W1lo, INdim, Hdim,  Hdim);
        dyn_tsplit_kernel<<<gw2, tb, 0, stream>>>(W2, W2hi, W2lo, Hdim,  Hdim,  Hdim);
        dyn_tsplit_kernel<<<gw3, tb, 0, stream>>>(W3, W3hi, W3lo, Hdim,  OUTdim, W3P);
    }

    // 3) three expert-bucketed WMMA GEMMs (B tiles via TDM)
    dim3 blk(256, 1, 1);
    dim3 g1(Bsz / BM, Hdim / BN, Edim);   // (64, 8, 8)
    dim3 g3(Bsz / BM, W3P / BN, Edim);    // (64, 3, 8)

    dyn_gemm_kernel<1><<<g1, blk, 0, stream>>>(counts, rowlist, state, action,
                                               nullptr, nullptr,
                                               W1hi, W1lo, Hdim, INdim, b1,
                                               h1hi, h1lo, nullptr, nullptr);
    dyn_gemm_kernel<2><<<g1, blk, 0, stream>>>(counts, rowlist, state, action,
                                               h1hi, h1lo,
                                               W2hi, W2lo, Hdim, Hdim, b2,
                                               h2hi, h2lo, nullptr, nullptr);
    dyn_gemm_kernel<3><<<g3, blk, 0, stream>>>(counts, rowlist, state, action,
                                               h2hi, h2lo,
                                               W3hi, W3lo, W3P, Hdim, b3,
                                               nullptr, nullptr, dnext, drew);
}
